// HSQuadricsModel_56848187130506
// MI455X (gfx1250) — matrix operand order repaired and checked by reading, not compiled, not run on hardware
//
#include <hip/hip_runtime.h>
#include <stdint.h>

// ---------------------------------------------------------------------------
// Problem constants (from reference): B=2048, dim=512, nq=64
// ---------------------------------------------------------------------------
#define BATCH 2048
#define DIM   512
#define NQ    64
#define BM    64          // rows of P per workgroup
#define BN    128         // e-columns per chunk
#define BK    32          // K-slice per WMMA step
#define NCHUNK (DIM / BN) // 4
#define NKSTEP (DIM / BK) // 16

typedef __attribute__((ext_vector_type(16))) __bf16 bf16x16;
typedef __attribute__((ext_vector_type(8)))  __bf16 bf16x8;
typedef __attribute__((ext_vector_type(8)))  float  v8f;
typedef __attribute__((ext_vector_type(4)))  unsigned int u32x4;
typedef __attribute__((ext_vector_type(8)))  int i32x8;
typedef __attribute__((ext_vector_type(4)))  int i32x4;

// ---------------------------------------------------------------------------
// Phase A: expand triu q_coefs into full symmetric scaled bf16 matrix
// W[k][e][d] = q_coefs[triu(d,e)*NQ + k] * (d==e ? 2 : sqrt(2))   (symmetric)
// ---------------------------------------------------------------------------
__global__ __launch_bounds__(256) void build_w_kernel(const float* __restrict__ qc,
                                                      __bf16* __restrict__ W) {
    int gid = blockIdx.x * 256 + threadIdx.x;   // gid = ((k*DIM)+e)*DIM + d
    int d = gid & (DIM - 1);
    int e = (gid >> 9) & (DIM - 1);
    int k = gid >> 18;
    int i = d < e ? d : e;
    int j = d < e ? e : d;
    int idx = i * DIM - (i * (i - 1)) / 2 + (j - i);   // row-major triu index
    float scale = (d == e) ? 2.0f : 1.41421356237f;
    float v = qc[idx * NQ + k] * scale;
    W[gid] = (__bf16)v;
}

__global__ __launch_bounds__(256) void conv_points_kernel(const float* __restrict__ p,
                                                          __bf16* __restrict__ p16) {
    int gid = blockIdx.x * 256 + threadIdx.x;
    p16[gid] = (__bf16)p[gid];
}

__global__ __launch_bounds__(256) void col_norms_kernel(const float* __restrict__ qc,
                                                        float* __restrict__ norms) {
    __shared__ float red[256];
    const int n_mono = DIM * (DIM - 1) / 2 + DIM;  // 131328
    int k = blockIdx.x, tid = threadIdx.x;
    float s = 0.0f;
    for (int r = tid; r < n_mono; r += 256) {
        float v = qc[r * NQ + k];
        s += v * v;
    }
    red[tid] = s;
    __syncthreads();
    for (int off = 128; off > 0; off >>= 1) {
        if (tid < off) red[tid] += red[tid + off];
        __syncthreads();
    }
    if (tid == 0) norms[k] = sqrtf(red[0]);
}

// ---------------------------------------------------------------------------
// TDM: 2D tile load Global -> LDS (bf16 elements, data_size code = 1)
// Descriptor layout per CDNA5 ISA 8.3/8.4 (groups 2/3 zero: 2D tensor).
// This toolchain's builtin takes 6 args: (g0, g1, g2, g3, g_extra, cpol).
// ---------------------------------------------------------------------------
__device__ __forceinline__ void tdm_load_2d(unsigned lds_off, const void* gptr,
                                            unsigned tile_d0, unsigned tile_d1,
                                            unsigned tensor_d0, unsigned tensor_d1,
                                            unsigned long long stride0) {
    unsigned long long ga = (unsigned long long)(uintptr_t)gptr;
    u32x4 g0;
    g0.x = 1u;                                            // count=1, user mode
    g0.y = lds_off;                                       // LDS byte address
    g0.z = (unsigned)(ga & 0xFFFFFFFFu);                  // global addr [31:0]
    g0.w = (unsigned)((ga >> 32) & 0x1FFFFFFu) | (2u << 30); // addr[56:32] | type=2
    i32x8 g1;
    g1[0] = (int)(1u << 16);                              // wg_mask=0, data_size=1 (2B)
    g1[1] = (int)((tensor_d0 & 0xFFFFu) << 16);           // tensor_dim0[15:0]
    g1[2] = (int)(((tensor_d0 >> 16) & 0xFFFFu) | ((tensor_d1 & 0xFFFFu) << 16));
    g1[3] = (int)(((tensor_d1 >> 16) & 0xFFFFu) | ((tile_d0 & 0xFFFFu) << 16));
    g1[4] = (int)(tile_d1 & 0xFFFFu);                     // tile_dim1, tile_dim2=0
    g1[5] = (int)(unsigned)(stride0 & 0xFFFFFFFFu);       // dim0_stride[31:0]
    g1[6] = (int)(unsigned)((stride0 >> 32) & 0xFFFFu);   // dim0_stride[47:32]
    g1[7] = 0;
    i32x4 z4 = {0, 0, 0, 0};
    i32x8 z8 = {0, 0, 0, 0, 0, 0, 0, 0};
    __builtin_amdgcn_tensor_load_to_lds(g0, g1, z4, z4, z8, 0);
}

// ---------------------------------------------------------------------------
// Main kernel: grid = (BATCH/BM, NQ). One quadric k, 64 batch rows per WG.
//   G = P_tile(64x512) @ M_k(512x512) [+ l]  via v_wmma_f32_16x16x32_bf16,
//   reduced on the fly into grad / p.G / p.l, then scored.
// ---------------------------------------------------------------------------
__global__ __launch_bounds__(256) void quadric_score_kernel(
        const __bf16* __restrict__ P16,      // [BATCH][DIM]
        const __bf16* __restrict__ W,        // [NQ][DIM(e)][DIM(d)]
        const float* __restrict__ l_coefs,   // [DIM][NQ]
        const float* __restrict__ free_c,    // [NQ]
        const float* __restrict__ norms,     // [NQ]
        float* __restrict__ out) {           // [BATCH][NQ]
    __shared__ __bf16 A_sh[BM * DIM];        // 64 KB: P tile (row-major, d contiguous)
    __shared__ __bf16 B_sh[2][BN * BK];      // 16 KB: double-buffered M_k slice [e][d]
    __shared__ float  l_sh[DIM];
    __shared__ float  grad_acc[BM], pg_acc[BM], pl_acc[BM];

    const int tid  = threadIdx.x;
    const int lane = tid & 31;
    const int wave = tid >> 5;
    const int l15  = lane & 15;
    const int half = lane >> 4;              // 0: M/K low half, 1: high half
    const int wm   = wave & 3;               // M-tile (16 rows each)
    const int wn   = wave >> 2;              // N-group (64 cols each)
    const int btile = blockIdx.x;
    const int k     = blockIdx.y;

    const __bf16* Wk = W + (size_t)k * DIM * DIM;

    // ---- stage l column, zero accumulators, kick off TDM loads ----
    for (int i = tid; i < DIM; i += 256) l_sh[i] = l_coefs[i * NQ + k];
    if (tid < BM) { grad_acc[tid] = 0.0f; pg_acc[tid] = 0.0f; pl_acc[tid] = 0.0f; }

    if (wave == 0) {
        // A tile: 64 consecutive rows x 512 -> one contiguous 32768-element tile
        tdm_load_2d((unsigned)(uintptr_t)&A_sh[0],
                    P16 + (size_t)btile * BM * DIM,
                    /*tile*/ BM * DIM, 1, /*tensor*/ BM * DIM, 1,
                    (unsigned long long)(BM * DIM));
        // first B slice: rows e=0..127, d=0..31
        tdm_load_2d((unsigned)(uintptr_t)&B_sh[0][0], Wk,
                    /*tile*/ BK, BN, /*tensor*/ DIM, DIM,
                    (unsigned long long)DIM);
    }

    float r_grad[8], r_pg[8], r_pl[8];
#pragma unroll
    for (int j = 0; j < 8; ++j) { r_grad[j] = 0.0f; r_pg[j] = 0.0f; r_pl[j] = 0.0f; }

    const v8f vzero = {};
    int step = 0;                            // global pipeline step = (ec/BN)*NKSTEP + kkIdx
    for (int ec = 0; ec < DIM; ec += BN) {
        v8f acc[4];
#pragma unroll
        for (int nt = 0; nt < 4; ++nt) acc[nt] = vzero;

        for (int kkIdx = 0; kkIdx < NKSTEP; ++kkIdx, ++step) {
            const int kk = kkIdx * BK;
            const int nstep = step + 1;
            const bool more = nstep < NCHUNK * NKSTEP;
            if (wave == 0) {
                if (more) {
                    int nec = (nstep >> 4) * BN;           // next e-chunk base
                    int nkk = (nstep & 15) * BK;           // next k-slice base
                    tdm_load_2d((unsigned)(uintptr_t)&B_sh[nstep & 1][0],
                                Wk + (size_t)nec * DIM + nkk,
                                BK, BN, DIM, DIM, (unsigned long long)DIM);
                    __builtin_amdgcn_s_wait_tensorcnt(1); // current slice (+A) done
                } else {
                    __builtin_amdgcn_s_wait_tensorcnt(0);
                }
            }
            __syncthreads();

            // A fragment: row = wm*16 + l15; lane half picks K {0..7,16..23}/{8..15,24..31}
            const __bf16* Ap = &A_sh[(wm * 16 + l15) * DIM + kk + half * 8];
            bf16x8 alo = *(const bf16x8*)Ap;
            bf16x8 ahi = *(const bf16x8*)(Ap + 16);
            bf16x16 afrag = __builtin_shufflevector(alo, ahi,
                0, 1, 2, 3, 4, 5, 6, 7, 8, 9, 10, 11, 12, 13, 14, 15);

            const __bf16* Bbase = &B_sh[step & 1][0];
#pragma unroll
            for (int nt = 0; nt < 4; ++nt) {
                int coll = wn * 64 + nt * 16 + l15;        // column within chunk
                const __bf16* Bp = Bbase + coll * BK + half * 8;
                bf16x8 blo = *(const bf16x8*)Bp;
                bf16x8 bhi = *(const bf16x8*)(Bp + 16);
                bf16x16 bfrag = __builtin_shufflevector(blo, bhi,
                    0, 1, 2, 3, 4, 5, 6, 7, 8, 9, 10, 11, 12, 13, 14, 15);
                acc[nt] = __builtin_amdgcn_wmma_f32_16x16x32_bf16(
                    false, afrag, false, bfrag, (short)0, acc[nt], false, false);
            }
            __syncthreads();                 // buffer reuse protection
        }

        // ---- fold l and reduce this e-chunk into per-row register accumulators ----
#pragma unroll
        for (int nt = 0; nt < 4; ++nt) {
            int e_col = ec + wn * 64 + nt * 16 + l15;
            float lv = l_sh[e_col];
#pragma unroll
            for (int j = 0; j < 8; ++j) {
                float g = acc[nt][j] + lv;                 // G = M p + l
                int rloc = wm * 16 + half * 8 + j;         // C layout: row=j(+8)
                float pv = (float)A_sh[rloc * DIM + e_col];
                r_grad[j] += g * g;
                r_pg[j]   += pv * g;
                r_pl[j]   += pv * lv;
            }
        }
    }

    // ---- cross-lane butterfly over the 16 lanes sharing each row ----
#pragma unroll
    for (int j = 0; j < 8; ++j) {
        for (int m = 1; m < 16; m <<= 1) {
            r_grad[j] += __shfl_xor(r_grad[j], m, 32);
            r_pg[j]   += __shfl_xor(r_pg[j], m, 32);
            r_pl[j]   += __shfl_xor(r_pl[j], m, 32);
        }
    }
    if (l15 == 0) {
#pragma unroll
        for (int j = 0; j < 8; ++j) {
            int row = wm * 16 + half * 8 + j;
            atomicAdd(&grad_acc[row], r_grad[j]);
            atomicAdd(&pg_acc[row],   r_pg[j]);
            atomicAdd(&pl_acc[row],   r_pl[j]);
        }
    }
    __syncthreads();

    // ---- final score: one thread per row ----
    if (tid < BM) {
        float gr = grad_acc[tid];
        float nk = norms[k];
        float val = fabsf(0.5f * pg_acc[tid] + 0.5f * pl_acc[tid] + free_c[k]);
        float sc = (sqrtf(gr * 0.25f + val * nk) - 0.5f * sqrtf(gr)) / nk;
        out[(size_t)(btile * BM + tid) * NQ + k] = sc;
    }
}

// ---------------------------------------------------------------------------
// Launcher. Inputs: points(B,dim) f32, q_coefs(n_mono,nq) f32,
//                   l_coefs(dim,nq) f32, free_coefs(1,nq) f32.
// Output: scores (B, nq) f32. Workspace: W bf16 (33.5MB) + P16 (2MB) + norms.
// ---------------------------------------------------------------------------
extern "C" void kernel_launch(void* const* d_in, const int* in_sizes, int n_in,
                              void* d_out, int out_size, void* d_ws, size_t ws_size,
                              hipStream_t stream) {
    const float* points = (const float*)d_in[0];
    const float* qc     = (const float*)d_in[1];
    const float* lc     = (const float*)d_in[2];
    const float* fc     = (const float*)d_in[3];
    float* out = (float*)d_out;

    char* ws = (char*)d_ws;
    __bf16* W    = (__bf16*)ws;                                   // NQ*DIM*DIM*2 = 33.5 MB
    __bf16* P16  = (__bf16*)(ws + (size_t)NQ * DIM * DIM * 2);    // BATCH*DIM*2 = 2 MB
    float*  norms = (float*)(ws + (size_t)NQ * DIM * DIM * 2 + (size_t)BATCH * DIM * 2);

    build_w_kernel<<<(NQ * DIM * DIM) / 256, 256, 0, stream>>>(qc, W);
    conv_points_kernel<<<(BATCH * DIM) / 256, 256, 0, stream>>>(points, P16);
    col_norms_kernel<<<NQ, 256, 0, stream>>>(qc, norms);

    dim3 grid(BATCH / BM, NQ);
    quadric_score_kernel<<<grid, 256, 0, stream>>>(P16, W, lc, fc, norms, out);
}
